// BlockWiseCrossAttention_89223650607346
// MI455X (gfx1250) — matrix operand order repaired and checked by reading, not compiled, not run on hardware
//
#include <hip/hip_runtime.h>

// ---------------------------------------------------------------------------
// Fused block-wise cross attention for MI455X (gfx1250, wave32, WMMA).
//   out1 = softmax(q1 k2^T / sqrt(E)) v2 ; out2 = softmax(q2 k1^T / sqrt(E)) v1
// One workgroup = one direction x one batch x 4 attention blocks (64 rows).
// All GEMMs via v_wmma_f32_16x16x32_bf16 (f32 accumulate).
// Phase 0: x -> bf16 in LDS.  Phase A: Q,K projections; barrier; V projection
// stored transposed over the dead x_q region.  Phase B: attention.
// Hot loop is software-pipelined: B-tile (global weights) double-buffered,
// all kR A-tiles preloaded so the 4 WMMAs issue back-to-back.
// ---------------------------------------------------------------------------

typedef __attribute__((ext_vector_type(16))) __bf16   v16bf;
typedef __attribute__((ext_vector_type(8)))  __bf16   bf16x8;
typedef __attribute__((ext_vector_type(4)))  __bf16   bf16x4;
typedef __attribute__((ext_vector_type(2)))  __bf16   bf16x2;
typedef __attribute__((ext_vector_type(8)))  float    v8f;
typedef __attribute__((ext_vector_type(4)))  float    f32x4;
typedef __attribute__((ext_vector_type(2)))  float    f32x2;
typedef __attribute__((ext_vector_type(8)))  unsigned v8u;

namespace {
constexpr int kB    = 8;
constexpr int kS    = 4096;
constexpr int kE    = 512;
constexpr int kBS   = 16;                   // attention block size
constexpr int kR    = 4;                    // attention blocks per WG (64 rows)
constexpr int kRows = kBS * kR;             // 64
constexpr int kSB   = (kS / kBS) / kR;      // 64 superblocks per batch
constexpr int LSTR  = 520;  // row stride (bf16): 1040B == 4 dwords mod 64 -> 16 rows hit 16 banks
constexpr int VSTR  = 72;   // transposed-V stride: 144B == 36 dwords mod 64 -> conflict-free, 16B aligned
constexpr int PSTR  = 24;   // P-scratch stride: 48B, 16B aligned, conflict-free
constexpr int kWaves   = 16;
constexpr int kThreads = kWaves * 32;

constexpr size_t kTileUS    = (size_t)kRows * LSTR;     // 33280 (x / q / k tile)
constexpr size_t kVRegionUS = (size_t)kE * VSTR;        // 36864 (transposed V, aliases x_q)
constexpr size_t kXqOff = 0;                            // x_q tile; later becomes V^T
constexpr size_t kXkOff = kVRegionUS;                   // x_kv tile
constexpr size_t kQOff  = kXkOff + kTileUS;
constexpr size_t kKOff  = kQOff + kTileUS;
constexpr size_t kPOff  = kKOff + kTileUS;
constexpr size_t kSmemUS    = kPOff + (size_t)kWaves * kBS * PSTR;
constexpr size_t kSmemBytes = kSmemUS * sizeof(unsigned short);   // ~286 KB < 320 KB WGP LDS
}

__device__ __forceinline__ unsigned cvt2(const float* p) {
  f32x2 f = *(const f32x2*)p;                        // 8B-aligned by construction
  bf16x2 b = __builtin_convertvector(f, bf16x2);     // v_cvt_pk_bf16_f32
  return __builtin_bit_cast(unsigned, b);
}

__device__ __forceinline__ v16bf concat8(bf16x8 lo, bf16x8 hi) {
  return __builtin_shufflevector(lo, hi, 0,1,2,3,4,5,6,7,8,9,10,11,12,13,14,15);
}

__device__ __forceinline__ v16bf load_btile(const float* wrow, int k0) {
  v8u buu;
  #pragma unroll
  for (int j = 0; j < 8; ++j) buu[j] = cvt2(wrow + k0 + 2 * j + ((j >> 2) << 3));
  return __builtin_bit_cast(v16bf, buu);
}

// One projection stream: 16-col output tile `ct` of  dst = src @ W^T + b.
// B-tile streamed from global weights (f32->bf16 in flight, double-buffered)
// and reused across kR row-tiles whose A operands come from LDS.
template <bool kTransposedStore>
__device__ __forceinline__ void proj_tile(const float* __restrict__ W,
                                          const float* __restrict__ bi,
                                          const __bf16* src, __bf16* dst,
                                          int ct, int l15, int half,
                                          int kbA, int kbB)
{
  const int n0 = ct << 4;
  const float*  wrow  = W + (size_t)(n0 + l15) * kE + kbB;
  const __bf16* arow0 = src + (size_t)l15 * LSTR + kbA;

  v16bf bmat = load_btile(wrow, 0);               // prologue: first B-tile
  v8f c[kR] = {};

  for (int kc = 0; kc < 16; ++kc) {
    const int k0 = kc * 32;

    // Preload next K-step's B-tile so global latency overlaps the WMMAs.
    v16bf bnext = bmat;
    if (kc < 15) {
      __builtin_prefetch(wrow + k0 + 96, 0, 3);   // global_prefetch_b8, 2 steps ahead
      bnext = load_btile(wrow, k0 + 32);
    }

    // Preload all kR A-tiles into distinct registers (one ds clause, one wait).
    v16bf amat[kR];
    #pragma unroll
    for (int rt = 0; rt < kR; ++rt) {
      const __bf16* ar = arow0 + (size_t)(rt * kBS) * LSTR + k0;
      amat[rt] = concat8(*(const bf16x8*)ar, *(const bf16x8*)(ar + 16));
    }

    #pragma unroll
    for (int rt = 0; rt < kR; ++rt)
      c[rt] = __builtin_amdgcn_wmma_f32_16x16x32_bf16(
                false, amat[rt], false, bmat, (short)0, c[rt], false, false);

    bmat = bnext;
  }

  const float bias = bi[n0 + l15];
  if constexpr (kTransposedStore) {
    // V stored transposed: dst[col][row]; this lane's 8 rows are contiguous.
    #pragma unroll
    for (int rt = 0; rt < kR; ++rt) {
      bf16x8 ob = __builtin_convertvector(c[rt] + bias, bf16x8);
      *(bf16x8*)(dst + (size_t)(n0 + l15) * VSTR + rt * kBS + half * 8) = ob;
    }
  } else {
    #pragma unroll
    for (int rt = 0; rt < kR; ++rt) {
      #pragma unroll
      for (int r = 0; r < 8; ++r)     // D layout: row = r + 8*half, col = l15
        dst[(size_t)(rt * kBS + r + half * 8) * LSTR + n0 + l15] = (__bf16)(c[rt][r] + bias);
    }
  }
}

__global__ __launch_bounds__(kThreads, 1)
void bwca_fused_wmma_kernel(const float* __restrict__ s1, const float* __restrict__ s2,
                            const float* __restrict__ Wq, const float* __restrict__ bq,
                            const float* __restrict__ Wk, const float* __restrict__ bk,
                            const float* __restrict__ Wv, const float* __restrict__ bv,
                            float* __restrict__ out)
{
  extern __shared__ __bf16 smem[];

  const int sb   = blockIdx.x;          // superblock 0..63
  const int bb   = blockIdx.y;          // batch 0..7
  const int dir  = blockIdx.z;          // 0: out1(q1,k2,v2)  1: out2(q2,k1,v1)
  const int row0 = sb * kRows;

  const float* xq  = (dir == 0) ? s1 : s2;
  const float* xkv = (dir == 0) ? s2 : s1;
  const float* xq_base  = xq  + ((size_t)bb * kS + row0) * kE;
  const float* xkv_base = xkv + ((size_t)bb * kS + row0) * kE;

  const int lane = threadIdx.x & 31;
  const int wave = threadIdx.x >> 5;
  const int half = lane >> 4;           // wave32 WMMA lane group
  const int l15  = lane & 15;
  const int kbA  = half ? 8  : 0;       // A-operand K base for this lane group
  const int kbB  = half ? 8 : 0;        // B-operand K base for this lane group

  // ================= Phase 0: stage x1/x2 tiles as bf16 in LDS =================
  // 2 tiles * 64 rows * 512 cols in chunks of 8 elements = 8192 chunks.
  for (int ch = threadIdx.x; ch < 8192; ch += kThreads) {
    const int t   = ch >> 12;           // 0: xq, 1: xkv
    const int rem = ch & 4095;
    const int row = rem >> 6;
    const int e0  = (rem & 63) << 3;
    const float* g = (t ? xkv_base : xq_base) + (size_t)row * kE + e0;
    f32x4 f0 = *(const f32x4*)g;
    f32x4 f1 = *(const f32x4*)(g + 4);
    bf16x4 b0 = __builtin_convertvector(f0, bf16x4);
    bf16x4 b1 = __builtin_convertvector(f1, bf16x4);
    bf16x8 b  = __builtin_shufflevector(b0, b1, 0,1,2,3,4,5,6,7);
    const size_t off = (t ? kXkOff : kXqOff) + (size_t)row * LSTR + e0;
    *(bf16x8*)(smem + off) = b;
  }
  __syncthreads();

  // ======================= Phase A: Q/K/V projections =========================
  // Q and K first (both x tiles live), barrier, then V transposed over x_q.
  for (int it = wave; it < 32; it += kWaves)
    proj_tile<false>(Wq, bq, smem + kXqOff, smem + kQOff, it, l15, half, kbA, kbB);
  for (int it = wave; it < 32; it += kWaves)
    proj_tile<false>(Wk, bk, smem + kXkOff, smem + kKOff, it, l15, half, kbA, kbB);
  __syncthreads();                      // x_q now dead -> V^T may overwrite it
  for (int it = wave; it < 32; it += kWaves)
    proj_tile<true>(Wv, bv, smem + kXkOff, smem + kXqOff, it, l15, half, kbA, kbB);
  __syncthreads();

  // ========================= Phase B: attention ===============================
  const int ab = wave >> 2;             // attention block 0..3
  const int qd = wave & 3;              // column quarter 0..3

  const __bf16* qrow = smem + kQOff + (size_t)(ab * kBS + l15) * LSTR + kbA;
  const __bf16* krow = smem + kKOff + (size_t)(ab * kBS + l15) * LSTR + kbB;

  v8f sAcc = {};
  for (int kc = 0; kc < 16; ++kc) {
    const int k0 = kc * 32;
    bf16x8 alo = *(const bf16x8*)(qrow + k0);
    bf16x8 ahi = *(const bf16x8*)(qrow + k0 + 16);
    bf16x8 blo = *(const bf16x8*)(krow + k0);       // B[e][n] = k[n][e]
    bf16x8 bhi = *(const bf16x8*)(krow + k0 + 16);
    sAcc = __builtin_amdgcn_wmma_f32_16x16x32_bf16(
             false, concat8(alo, ahi), false, concat8(blo, bhi),
             (short)0, sAcc, false, false);
  }

  // Row softmax: rows live across the 16 lanes of each half-group.
  const float inv_scale = 4.4194173824159216e-2f;   // 1/sqrt(512)
  __bf16* myP = smem + kPOff + (size_t)wave * kBS * PSTR;
  #pragma unroll
  for (int r = 0; r < 8; ++r) {
    float v = sAcc[r] * inv_scale;
    float m = v;
    m = fmaxf(m, __shfl_xor(m, 1));
    m = fmaxf(m, __shfl_xor(m, 2));
    m = fmaxf(m, __shfl_xor(m, 4));
    m = fmaxf(m, __shfl_xor(m, 8));
    float pe  = __expf(v - m);
    float sum = pe;
    sum += __shfl_xor(sum, 1);
    sum += __shfl_xor(sum, 2);
    sum += __shfl_xor(sum, 4);
    sum += __shfl_xor(sum, 8);
    myP[(r + half * 8) * PSTR + l15] = (__bf16)(pe / sum);    // D-layout store
  }
  // Reload P in A layout (same-wave LDS ops are ordered); K=16..31 zero-padded.
  bf16x8 plo = *(const bf16x8*)(myP + (size_t)l15 * PSTR + kbA);
  bf16x8 z8  = {};
  const v16bf pa = concat8(plo, z8);

  float* obase = out + (size_t)dir * kB * kS * kE
               + ((size_t)bb * kS + row0 + ab * kBS) * kE;

  const v16bf zv = {};
  for (int i = 0; i < 8; ++i) {
    const int n0 = (qd * 8 + i) * 16;
    // B[k][n] = v[k][n0+n]: elements 0..7 hold k = 8*half + s, elements 8..15 -> 0.
    const __bf16* vcol = smem + kXqOff + (size_t)(n0 + l15) * VSTR + ab * kBS;
    bf16x8 v0 = *(const bf16x8*)vcol;
    bf16x8 v1 = *(const bf16x8*)(vcol + 8);
    v16bf bmat = concat8(half ? v1 : v0, z8);
    v8f o = {};
    o = __builtin_amdgcn_wmma_f32_16x16x32_bf16(
          false, pa, false, bmat, (short)0, o, false, false);
    #pragma unroll
    for (int r = 0; r < 8; ++r)
      obase[(size_t)(r + half * 8) * kE + n0 + l15] = o[r];
  }
}

extern "C" void kernel_launch(void* const* d_in, const int* in_sizes, int n_in,
                              void* d_out, int out_size, void* d_ws, size_t ws_size,
                              hipStream_t stream) {
  (void)in_sizes; (void)n_in; (void)out_size; (void)d_ws; (void)ws_size;
  const float* s1 = (const float*)d_in[0];
  const float* s2 = (const float*)d_in[1];
  const float* Wq = (const float*)d_in[2];
  const float* bq = (const float*)d_in[3];
  const float* Wk = (const float*)d_in[4];
  const float* bk = (const float*)d_in[5];
  const float* Wv = (const float*)d_in[6];
  const float* bv = (const float*)d_in[7];
  float* out = (float*)d_out;

  // Opt in to >64KB dynamic LDS (gfx1250 WGP has 320 KB). Not a stream op.
  (void)hipFuncSetAttribute((const void*)bwca_fused_wmma_kernel,
                            hipFuncAttributeMaxDynamicSharedMemorySize,
                            (int)kSmemBytes);

  dim3 grid(kSB, kB, 2);        // 64 superblocks x 8 batches x 2 directions
  bwca_fused_wmma_kernel<<<grid, dim3(kThreads), kSmemBytes, stream>>>(
      s1, s2, Wq, bq, Wk, bk, Wv, bv, out);
}
